// GraphVQAEncoder_22376779612492
// MI455X (gfx1250) — matrix operand order repaired and checked by reading, not compiled, not run on hardware
//
#include <hip/hip_runtime.h>

#define F 128

typedef __attribute__((ext_vector_type(16))) __bf16 v16bf;
typedef __attribute__((ext_vector_type(8)))  __bf16 v8bf;
typedef __attribute__((ext_vector_type(8)))  float  v8f;

// ---------------- fragment helpers (CDNA5 wave32 WMMA layouts) ----------------
// A (16x32 bf16): lane holds row r=lane&15; K elems j0..7 -> hi*8+j, j8..15 -> 16+hi*8+j
__device__ __forceinline__ v16bf a_from_global(const float* row32, int lane, float scale) {
  const int hi = (lane >> 4) & 1;
  const float4* p = (const float4*)(row32 + hi * 8);
  const float4* q = (const float4*)(row32 + 16 + hi * 8);
  float4 a0 = p[0], a1 = p[1], b0 = q[0], b1 = q[1];
  v16bf a;
  a[0]=(__bf16)(a0.x*scale); a[1]=(__bf16)(a0.y*scale); a[2]=(__bf16)(a0.z*scale); a[3]=(__bf16)(a0.w*scale);
  a[4]=(__bf16)(a1.x*scale); a[5]=(__bf16)(a1.y*scale); a[6]=(__bf16)(a1.z*scale); a[7]=(__bf16)(a1.w*scale);
  a[8]=(__bf16)(b0.x*scale); a[9]=(__bf16)(b0.y*scale); a[10]=(__bf16)(b0.z*scale); a[11]=(__bf16)(b0.w*scale);
  a[12]=(__bf16)(b1.x*scale); a[13]=(__bf16)(b1.y*scale); a[14]=(__bf16)(b1.z*scale); a[15]=(__bf16)(b1.w*scale);
  return a;
}

__device__ __forceinline__ v16bf a_from_prod(const float* x32, const float* y32, int lane) {
  const int hi = (lane >> 4) & 1;
  const float4* px = (const float4*)(x32 + hi * 8);
  const float4* qx = (const float4*)(x32 + 16 + hi * 8);
  const float4* py = (const float4*)(y32 + hi * 8);
  const float4* qy = (const float4*)(y32 + 16 + hi * 8);
  float4 a0 = px[0], a1 = px[1], b0 = qx[0], b1 = qx[1];
  float4 c0 = py[0], c1 = py[1], d0 = qy[0], d1 = qy[1];
  v16bf a;
  a[0]=(__bf16)(a0.x*c0.x); a[1]=(__bf16)(a0.y*c0.y); a[2]=(__bf16)(a0.z*c0.z); a[3]=(__bf16)(a0.w*c0.w);
  a[4]=(__bf16)(a1.x*c1.x); a[5]=(__bf16)(a1.y*c1.y); a[6]=(__bf16)(a1.z*c1.z); a[7]=(__bf16)(a1.w*c1.w);
  a[8]=(__bf16)(b0.x*d0.x); a[9]=(__bf16)(b0.y*d0.y); a[10]=(__bf16)(b0.z*d0.z); a[11]=(__bf16)(b0.w*d0.w);
  a[12]=(__bf16)(b1.x*d1.x); a[13]=(__bf16)(b1.y*d1.y); a[14]=(__bf16)(b1.z*d1.z); a[15]=(__bf16)(b1.w*d1.w);
  return a;
}

__device__ __forceinline__ v16bf a_from_lds(const __bf16* tile, int lane, int kb) {
  const int r = lane & 15, hi = (lane >> 4) & 1;
  const __bf16* p = tile + r * F + kb + hi * 8;
  v8bf lo = *(const v8bf*)p;
  v8bf hh = *(const v8bf*)(p + 16);
  v16bf a;
#pragma unroll
  for (int j = 0; j < 8; ++j) { a[j] = lo[j]; a[8 + j] = hh[j]; }
  return a;
}

// B (32x16 bf16) from K-major (transposed) weight: lane holds col n=lane&15;
// K elems j0..7 -> hi*8+j, j8..15 -> 16+hi*8+j (two 16B loads, matches A).
__device__ __forceinline__ v16bf b_from_wt(const __bf16* Wt, int ldk, int n, int kb, int lane) {
  const __bf16* p = Wt + (size_t)n * ldk + kb + ((lane >> 4) & 1) * 8;
  v8bf lo = *(const v8bf*)p;
  v8bf hh = *(const v8bf*)(p + 16);
  v16bf b;
#pragma unroll
  for (int j = 0; j < 8; ++j) { b[j] = lo[j]; b[8 + j] = hh[j]; }
  return b;
}

__device__ __forceinline__ void zero_acc(v8f acc[8]) {
#pragma unroll
  for (int t = 0; t < 8; ++t)
#pragma unroll
    for (int j = 0; j < 8; ++j) acc[t][j] = 0.f;
}

// one K=32 step against all 8 N-tiles (output width 128)
__device__ __forceinline__ void wmma_step(v8f acc[8], v16bf a, const __bf16* Wt, int ldk, int kb, int lane) {
  const int r = lane & 15;
#pragma unroll
  for (int t = 0; t < 8; ++t) {
    v16bf b = b_from_wt(Wt, ldk, t * 16 + r, kb, lane);
    acc[t] = __builtin_amdgcn_wmma_f32_16x16x32_bf16(false, a, false, b, (short)0, acc[t], false, false);
  }
}

// D-layout (lane holds col n=lane&15, rows v+8*hi) -> row-major bf16 LDS tile (+bias,+relu)
__device__ __forceinline__ void bias_act_to_lds(__bf16* T, const v8f acc[8], const float* bias, int lane, bool relu) {
  const int r = lane & 15, hi = (lane >> 4) & 1;
#pragma unroll
  for (int t = 0; t < 8; ++t) {
    float bv = bias[t * 16 + r];
#pragma unroll
    for (int v = 0; v < 8; ++v) {
      float h = acc[t][v] + bv;
      if (relu) h = fmaxf(h, 0.f);
      T[(v + 8 * hi) * F + t * 16 + r] = (__bf16)h;
    }
  }
  __asm__ volatile("s_wait_dscnt 0x0" ::: "memory"); // wave-private tile: split-counter wait, no barrier
}

__device__ __forceinline__ unsigned fenc(float f) {
  unsigned u = __float_as_uint(f);
  return (u & 0x80000000u) ? ~u : (u | 0x80000000u);
}
__device__ __forceinline__ float fdec(unsigned u) {
  return (u & 0x80000000u) ? __uint_as_float(u & 0x7FFFFFFFu) : __uint_as_float(~u);
}

// ---------------- small utility kernels ----------------
__global__ void fill_f32(float* p, float v, long n) {
  long i = (long)blockIdx.x * 256 + threadIdx.x;
  if (i < n) p[i] = v;
}
__global__ void fill_u32(unsigned* p, unsigned v, int n) {
  int i = blockIdx.x * 256 + threadIdx.x;
  if (i < n) p[i] = v;
}
// W [K,N] f32 -> Wt [N,K] bf16
__global__ void wprep_kernel(const float* __restrict__ W, __bf16* __restrict__ Wt, int K, int Nout) {
  int i = blockIdx.x * 256 + threadIdx.x;
  if (i < K * Nout) {
    int k = i / Nout, n = i % Nout;
    Wt[(size_t)n * K + k] = (__bf16)W[i];
  }
}

// ---------------- edge pipeline: e = MLP(src|dst|ea); msg = MLP(src|e); scatter ----------------
__global__ void __launch_bounds__(256) edge_kernel(
    const float* __restrict__ x, const float* __restrict__ ea, const int* __restrict__ eidx,
    const __bf16* We1t, const float* be1, const __bf16* We2t, const float* be2,
    const __bf16* Wn1t, const float* bn1, const __bf16* Wn1bt, const float* bn1b,
    float* e_out, float* sums, float* cnt, int E, int ntiles) {
  __shared__ __align__(32) __bf16 lds[8][16 * F];
  const int wave = threadIdx.x >> 5, lane = threadIdx.x & 31;
  const int tile = blockIdx.x * 8 + wave;
  if (tile >= ntiles) return;
  const int r = lane & 15, hi = (lane >> 4) & 1;
  const int ebase = tile * 16;
  const int erow = ebase + r;
  const int srow = eidx[erow];       // source node of this lane's edge
  const int crow = eidx[E + erow];   // dest node
  const float* srcp = x + (size_t)srow * F;
  const float* dstp = x + (size_t)crow * F;
  const float* eap  = ea + (size_t)erow * F;
  __bf16* T = lds[wave];
  v8f acc[8];

  // GEMM1: h1 = relu(concat(src,dst,ea)[16x384] @ We1 + be1)
  zero_acc(acc);
  for (int ks = 0; ks < 12; ++ks) {
    int kb = ks * 32;
    const float* ap = (kb < 128) ? (srcp + kb) : (kb < 256) ? (dstp + kb - 128) : (eap + kb - 256);
    v16bf a = a_from_global(ap, lane, 1.f);
    wmma_step(acc, a, We1t, 384, kb, lane);
  }
  bias_act_to_lds(T, acc, be1, lane, true);

  // GEMM2: e = h1 @ We2 + be2 ; write e (f32) + keep bf16 copy in T
  zero_acc(acc);
  for (int ks = 0; ks < 4; ++ks) {
    v16bf a = a_from_lds(T, lane, ks * 32);
    wmma_step(acc, a, We2t, 128, ks * 32, lane);
  }
#pragma unroll
  for (int t = 0; t < 8; ++t) {
    float bv = be2[t * 16 + r];
#pragma unroll
    for (int v = 0; v < 8; ++v) {
      float ev = acc[t][v] + bv;
      e_out[(size_t)(ebase + v + 8 * hi) * F + t * 16 + r] = ev;
      T[(v + 8 * hi) * F + t * 16 + r] = (__bf16)ev;
    }
  }
  __asm__ volatile("s_wait_dscnt 0x0" ::: "memory");

  // GEMM3: h2 = relu(concat(src,e)[16x256] @ Wn1 + bn1)
  zero_acc(acc);
  for (int ks = 0; ks < 8; ++ks) {
    int kb = ks * 32;
    v16bf a = (kb < 128) ? a_from_global(srcp + kb, lane, 1.f) : a_from_lds(T, lane, kb - 128);
    wmma_step(acc, a, Wn1t, 256, kb, lane);
  }
  bias_act_to_lds(T, acc, bn1, lane, true);

  // GEMM4: msg = h2 @ Wn1b + bn1b ; scatter-add into sums[dest]
  zero_acc(acc);
  for (int ks = 0; ks < 4; ++ks) {
    v16bf a = a_from_lds(T, lane, ks * 32);
    wmma_step(acc, a, Wn1bt, 128, ks * 32, lane);
  }
#pragma unroll
  for (int t = 0; t < 8; ++t) {
    float bv = bn1b[t * 16 + r];
#pragma unroll
    for (int v = 0; v < 8; ++v) {
      int dest = eidx[E + ebase + v + 8 * hi];  // uniform across the 16-lane group
      atomicAdd(&sums[(size_t)dest * F + t * 16 + r], acc[t][v] + bv);
    }
  }
  if (hi == 0) atomicAdd(&cnt[crow], 1.0f);
}

// ---------------- node pipeline: xo = MLP(x|agg); xn = MLP(xo) ----------------
__global__ void __launch_bounds__(256) node_kernel(
    const float* __restrict__ x, const float* sums, const float* __restrict__ cnt,
    const __bf16* Wn2t, const float* bn2, const __bf16* Wn2bt, const float* bn2b,
    const __bf16* Wv1t, const float* bv1, const __bf16* Wv2t, const float* bv2,
    float* xo_out, float* xn, int ntiles) {
  __shared__ __align__(32) __bf16 lds[8][16 * F];
  const int wave = threadIdx.x >> 5, lane = threadIdx.x & 31;
  const int tile = blockIdx.x * 8 + wave;
  if (tile >= ntiles) return;
  const int r = lane & 15, hi = (lane >> 4) & 1;
  const int nbase = tile * 16;
  const int node = nbase + r;
  const float* xp = x + (size_t)node * F;
  const float* sp = sums + (size_t)node * F;
  const float inv = 1.0f / fmaxf(cnt[node], 1.0f);  // scatter_mean divisor
  __bf16* T = lds[wave];
  v8f acc[8];

  // GEMM1: h = relu(concat(x, agg)[16x256] @ Wn2 + bn2), agg = sums*inv folded into cvt
  zero_acc(acc);
  for (int ks = 0; ks < 8; ++ks) {
    int kb = ks * 32;
    v16bf a = (kb < 128) ? a_from_global(xp + kb, lane, 1.f)
                         : a_from_global(sp + kb - 128, lane, inv);
    wmma_step(acc, a, Wn2t, 256, kb, lane);
  }
  bias_act_to_lds(T, acc, bn2, lane, true);

  // GEMM2: xo = h @ Wn2b + bn2b ; write xo f32 + bf16 copy
  zero_acc(acc);
  for (int ks = 0; ks < 4; ++ks) {
    v16bf a = a_from_lds(T, lane, ks * 32);
    wmma_step(acc, a, Wn2bt, 128, ks * 32, lane);
  }
#pragma unroll
  for (int t = 0; t < 8; ++t) {
    float bv = bn2b[t * 16 + r];
#pragma unroll
    for (int v = 0; v < 8; ++v) {
      float xo = acc[t][v] + bv;
      xo_out[(size_t)(nbase + v + 8 * hi) * F + t * 16 + r] = xo;
      T[(v + 8 * hi) * F + t * 16 + r] = (__bf16)xo;
    }
  }
  __asm__ volatile("s_wait_dscnt 0x0" ::: "memory");

  // GEMM3: h = relu(xo @ Wv1 + bv1)
  zero_acc(acc);
  for (int ks = 0; ks < 4; ++ks) {
    v16bf a = a_from_lds(T, lane, ks * 32);
    wmma_step(acc, a, Wv1t, 128, ks * 32, lane);
  }
  bias_act_to_lds(T, acc, bv1, lane, true);

  // GEMM4: xn = h @ Wv2 + bv2 (xn aliases sums; this wave's rows were fully read in GEMM1)
  zero_acc(acc);
  for (int ks = 0; ks < 4; ++ks) {
    v16bf a = a_from_lds(T, lane, ks * 32);
    wmma_step(acc, a, Wv2t, 128, ks * 32, lane);
  }
#pragma unroll
  for (int t = 0; t < 8; ++t) {
    float bv = bv2[t * 16 + r];
#pragma unroll
    for (int v = 0; v < 8; ++v)
      xn[(size_t)(nbase + v + 8 * hi) * F + t * 16 + r] = acc[t][v] + bv;
  }
}

// ---------------- q = MLP(u): 64x128, one block ----------------
__global__ void __launch_bounds__(128) q_kernel(
    const float* __restrict__ u, const __bf16* Wq1t, const float* bq1,
    const __bf16* Wq2t, const float* bq2, float* q) {
  __shared__ __align__(32) __bf16 lds[4][16 * F];
  const int wave = threadIdx.x >> 5, lane = threadIdx.x & 31;
  const int r = lane & 15, hi = (lane >> 4) & 1;
  const float* up = u + (size_t)(wave * 16 + r) * F;
  __bf16* T = lds[wave];
  v8f acc[8];
  zero_acc(acc);
  for (int ks = 0; ks < 4; ++ks) {
    v16bf a = a_from_global(up + ks * 32, lane, 1.f);
    wmma_step(acc, a, Wq1t, 128, ks * 32, lane);
  }
  bias_act_to_lds(T, acc, bq1, lane, true);
  zero_acc(acc);
  for (int ks = 0; ks < 4; ++ks) {
    v16bf a = a_from_lds(T, lane, ks * 32);
    wmma_step(acc, a, Wq2t, 128, ks * 32, lane);
  }
#pragma unroll
  for (int t = 0; t < 8; ++t) {
    float bv = bq2[t * 16 + r];
#pragma unroll
    for (int v = 0; v < 8; ++v)
      q[(size_t)(wave * 16 + v + 8 * hi) * F + t * 16 + r] = acc[t][v] + bv;
  }
}

// ---------------- gate = MLP(q[batch]*xn) -> [N,1]; seeded segment max ----------------
__global__ void __launch_bounds__(256) gate_kernel(
    const float* __restrict__ xn, const float* __restrict__ q, const int* __restrict__ batch,
    const __bf16* Wg1t, const float* bg1, const float* __restrict__ Wg2, const float* bg2,
    float* gate, unsigned* gmaxu, int ntiles) {
  const int wave = threadIdx.x >> 5, lane = threadIdx.x & 31;
  const int tile = blockIdx.x * 8 + wave;
  if (tile >= ntiles) return;
  const int r = lane & 15, hi = (lane >> 4) & 1;
  const int node = tile * 16 + r;
  const float* qp = q + (size_t)batch[node] * F;
  const float* xp = xn + (size_t)node * F;
  v8f acc[8];
  zero_acc(acc);
  for (int ks = 0; ks < 4; ++ks) {
    v16bf a = a_from_prod(xp + ks * 32, qp + ks * 32, lane);
    wmma_step(acc, a, Wg1t, 128, ks * 32, lane);
  }
  // gate = relu(hg) . Wg2 + bg2   (dot over the 128 columns, butterfly across 16 lanes)
  float part[8];
#pragma unroll
  for (int v = 0; v < 8; ++v) part[v] = 0.f;
#pragma unroll
  for (int t = 0; t < 8; ++t) {
    float bv = bg1[t * 16 + r];
    float w  = Wg2[t * 16 + r];
#pragma unroll
    for (int v = 0; v < 8; ++v) part[v] += fmaxf(acc[t][v] + bv, 0.f) * w;
  }
#pragma unroll
  for (int v = 0; v < 8; ++v)
#pragma unroll
    for (int m = 1; m < 16; m <<= 1) part[v] += __shfl_xor(part[v], m, 16);
#pragma unroll
  for (int v = 0; v < 8; ++v) {
    if (r == v) {
      int nd = tile * 16 + v + 8 * hi;
      float gt = part[v] + bg2[0];
      gate[nd] = gt;
      atomicMax(gmaxu + batch[nd], fenc(gt));
    }
  }
}

__global__ void expden_kernel(float* gate, const int* __restrict__ batch,
                              const unsigned* __restrict__ gmaxu, float* den, int N) {
  int i = blockIdx.x * 256 + threadIdx.x;
  if (i >= N) return;
  int b = batch[i];
  float ev = __expf(gate[i] - fdec(gmaxu[b]));
  gate[i] = ev;  // gate buffer now holds gexp
  atomicAdd(den + b, ev);
}

// pooled[b] += alpha*xn : sorted batch -> run-length compressed atomic flush
__global__ void __launch_bounds__(128) pooled_kernel(
    const float* __restrict__ gexp, const float* __restrict__ den, const float* __restrict__ xn,
    const int* __restrict__ batch, float* pooled, int N) {
  const int c = threadIdx.x;
  const int base = blockIdx.x * 32;
  float acc = 0.f;
  int cur = -1;
  for (int k = 0; k < 32; ++k) {
    int nd = base + k;
    if (nd >= N) break;
    int b = batch[nd];
    if (b != cur) {
      if (cur >= 0) atomicAdd(&pooled[(size_t)cur * F + c], acc);
      acc = 0.f; cur = b;
    }
    float alpha = gexp[nd] / (den[b] + 1e-16f);
    acc += alpha * xn[(size_t)nd * F + c];
  }
  if (cur >= 0) atomicAdd(&pooled[(size_t)cur * F + c], acc);
}

// ---------------- host orchestration ----------------
extern "C" void kernel_launch(void* const* d_in, const int* in_sizes, int n_in,
                              void* d_out, int out_size, void* d_ws, size_t ws_size,
                              hipStream_t stream) {
  const float* x  = (const float*)d_in[0];
  const float* ea = (const float*)d_in[1];
  const float* u  = (const float*)d_in[2];
  const int* eidx = (const int*)d_in[3];
  const int* batch = (const int*)d_in[4];
  const int N = in_sizes[0] / F;
  const int E = in_sizes[1] / F;
  const int B = in_sizes[2] / F;

  // workspace carve-out
  size_t off = 0;
  auto take = [&](size_t bytes) -> char* {
    char* p = (char*)d_ws + off;
    off += (bytes + 255) & ~(size_t)255;
    return p;
  };
  __bf16* We1t  = (__bf16*)take(384 * 128 * 2);
  __bf16* We2t  = (__bf16*)take(128 * 128 * 2);
  __bf16* Wn1t  = (__bf16*)take(256 * 128 * 2);
  __bf16* Wn1bt = (__bf16*)take(128 * 128 * 2);
  __bf16* Wn2t  = (__bf16*)take(256 * 128 * 2);
  __bf16* Wn2bt = (__bf16*)take(128 * 128 * 2);
  __bf16* Wv1t  = (__bf16*)take(128 * 128 * 2);
  __bf16* Wv2t  = (__bf16*)take(128 * 128 * 2);
  __bf16* Wq1t  = (__bf16*)take(128 * 128 * 2);
  __bf16* Wq2t  = (__bf16*)take(128 * 128 * 2);
  __bf16* Wg1t  = (__bf16*)take(128 * 128 * 2);
  float* sums = (float*)take((size_t)N * F * 4);  // reused as xn after node_kernel
  float* cnt  = (float*)take((size_t)N * 4);
  float* gate = (float*)take((size_t)N * 4);
  float* qws  = (float*)take((size_t)B * F * 4);
  unsigned* gmaxu = (unsigned*)take((size_t)B * 4);
  float* den  = (float*)take((size_t)B * 4);
  (void)ws_size; (void)n_in; (void)out_size;

  float* out = (float*)d_out;
  float* xo_out = out;
  float* e_out  = out + (size_t)N * F;
  float* pooled = e_out + (size_t)E * F;

  // zero/seed accumulators
  long nsum = (long)N * F;
  fill_f32<<<dim3((nsum + 255) / 256), dim3(256), 0, stream>>>(sums, 0.f, nsum);
  fill_f32<<<dim3((N + 255) / 256), dim3(256), 0, stream>>>(cnt, 0.f, N);
  fill_f32<<<dim3((B + 255) / 256), dim3(256), 0, stream>>>(den, 0.f, B);
  fill_f32<<<dim3((B * F + 255) / 256), dim3(256), 0, stream>>>(pooled, 0.f, B * F);
  fill_u32<<<dim3((B + 255) / 256), dim3(256), 0, stream>>>(gmaxu, 0u, B);

  // weights -> transposed bf16
  auto WP = [&](int idx, __bf16* dst, int K, int Nout) {
    int n = K * Nout;
    wprep_kernel<<<dim3((n + 255) / 256), dim3(256), 0, stream>>>((const float*)d_in[idx], dst, K, Nout);
  };
  WP(5, We1t, 384, 128);  WP(7, We2t, 128, 128);
  WP(9, Wn1t, 256, 128);  WP(11, Wn1bt, 128, 128);
  WP(13, Wn2t, 256, 128); WP(15, Wn2bt, 128, 128);
  WP(17, Wv1t, 128, 128); WP(19, Wv2t, 128, 128);
  WP(21, Wq1t, 128, 128); WP(23, Wq2t, 128, 128);
  WP(25, Wg1t, 128, 128);

  q_kernel<<<dim3(1), dim3(128), 0, stream>>>(u, Wq1t, (const float*)d_in[22],
                                              Wq2t, (const float*)d_in[24], qws);

  const int etiles = (E + 15) / 16;
  edge_kernel<<<dim3((etiles + 7) / 8), dim3(256), 0, stream>>>(
      x, ea, eidx,
      We1t, (const float*)d_in[6], We2t, (const float*)d_in[8],
      Wn1t, (const float*)d_in[10], Wn1bt, (const float*)d_in[12],
      e_out, sums, cnt, E, etiles);

  const int ntiles = (N + 15) / 16;
  node_kernel<<<dim3((ntiles + 7) / 8), dim3(256), 0, stream>>>(
      x, sums, cnt,
      Wn2t, (const float*)d_in[14], Wn2bt, (const float*)d_in[16],
      Wv1t, (const float*)d_in[18], Wv2t, (const float*)d_in[20],
      xo_out, sums /*xn alias*/, ntiles);

  gate_kernel<<<dim3((ntiles + 7) / 8), dim3(256), 0, stream>>>(
      sums /*xn*/, qws, batch,
      Wg1t, (const float*)d_in[26], (const float*)d_in[27], (const float*)d_in[28],
      gate, gmaxu, ntiles);

  expden_kernel<<<dim3((N + 255) / 256), dim3(256), 0, stream>>>(gate, batch, gmaxu, den, N);

  pooled_kernel<<<dim3((N + 31) / 32), dim3(128), 0, stream>>>(gate, den, sums /*xn*/, batch, pooled, N);
}
